// PointTransformerLayer_75419625718654
// MI455X (gfx1250) — matrix operand order repaired and checked by reading, not compiled, not run on hardware
//
#include <hip/hip_runtime.h>

// ---------------------------------------------------------------------------
// Point Transformer layer for gfx1250 (MI455X), bf16 WMMA everywhere.
// B=2, N=4096, DIM=512, HEADS=8, HEAD_DIM=64, MLP_HID=2048.
// Flash attention stages K/V tiles in LDS via global_load_async_to_lds_b128
// with double buffering driven by s_wait_asynccnt.
// ---------------------------------------------------------------------------

typedef __attribute__((ext_vector_type(16))) __bf16 v16bf;
typedef __attribute__((ext_vector_type(8)))  __bf16 v8bf;
typedef __attribute__((ext_vector_type(8)))  float  v8f;

#define BN    8192      // B*N rows
#define NTOK  4096
#define DIMC  512
#define HD    64
#define BHN   16        // B*HEADS
#define HIDN  2048
#define SCALE_Q 0.125f  // 64^-0.5

#define WMMA_BF16(a, b, c) \
  __builtin_amdgcn_wmma_f32_16x16x32_bf16(false, (a), false, (b), (short)0, (c), false, false)

__device__ __forceinline__ unsigned short f2bf(float f) {
  unsigned u = __float_as_uint(f);
  u += 0x7fffu + ((u >> 16) & 1u);          // round-to-nearest-even
  return (unsigned short)(u >> 16);
}
__device__ __forceinline__ float bf2f(unsigned short s) {
  return __uint_as_float(((unsigned)s) << 16);
}

// A-fragment (16x32 bf16): lane holds row lane%16; K chunks {8g..8g+7, 16+8g..}
__device__ __forceinline__ v16bf load_fragA(const unsigned short* p, int g) {
  union { v16bf v; v8bf h[2]; } u;
  u.h[0] = *(const v8bf*)(p + g * 8);
  u.h[1] = *(const v8bf*)(p + 16 + g * 8);
  return u.v;
}
// B-fragment (32x16 bf16): lane holds column lane%16; K chunks {8g..8g+7, 16+8g..}
__device__ __forceinline__ v16bf load_fragB(const unsigned short* p, int g) {
  union { v16bf v; v8bf h[2]; } u;
  u.h[0] = *(const v8bf*)(p + g * 8);
  u.h[1] = *(const v8bf*)(p + 16 + g * 8);
  return u.v;
}

// Async global->LDS copy (CDNA5). lds_off = workgroup LDS byte offset
// (low 32 bits of the generic pointer to a __shared__ object).
__device__ __forceinline__ void async_copy_b128(unsigned lds_off,
                                                const unsigned short* gptr) {
  asm volatile("global_load_async_to_lds_b128 %0, %1, off"
               :: "v"(lds_off), "v"(gptr) : "memory");
}

__device__ __forceinline__ float gelu_exact(float x) {
  return 0.5f * x * (1.0f + erff(x * 0.70710678118654752440f));
}

// ---------------------------------------------------------------------------
// Weight transpose + fp32 -> bf16:  src[R,C] -> dst[C,R]
// ---------------------------------------------------------------------------
__global__ void pt_wt_transpose_bf16(const float* __restrict__ src,
                                     unsigned short* __restrict__ dst,
                                     int R, int C) {
  int idx = blockIdx.x * blockDim.x + threadIdx.x;
  if (idx >= R * C) return;
  int r = idx / C, c = idx % C;
  dst[(size_t)c * R + r] = f2bf(src[idx]);
}

// ---------------------------------------------------------------------------
// LayerNorm over 512, fp32 in -> bf16 out. One 128-thread block per row.
// ---------------------------------------------------------------------------
__global__ __launch_bounds__(128) void pt_ln_bf16(const float* __restrict__ x,
                                                  const float* __restrict__ gw,
                                                  const float* __restrict__ bw,
                                                  unsigned short* __restrict__ out) {
  __shared__ float s1[128], s2[128];
  int row = blockIdx.x;
  int t = threadIdx.x;
  const float* xr = x + (size_t)row * DIMC;
  float4 v = ((const float4*)xr)[t];
  s1[t] = v.x + v.y + v.z + v.w;
  s2[t] = v.x * v.x + v.y * v.y + v.z * v.z + v.w * v.w;
  __syncthreads();
  for (int off = 64; off > 0; off >>= 1) {
    if (t < off) { s1[t] += s1[t + off]; s2[t] += s2[t + off]; }
    __syncthreads();
  }
  float mu = s1[0] * (1.0f / DIMC);
  float var = s2[0] * (1.0f / DIMC) - mu * mu;
  float rs = rsqrtf(var + 1e-5f);
  int c = t * 4;
  unsigned short* o = out + (size_t)row * DIMC + c;
  o[0] = f2bf((v.x - mu) * rs * gw[c + 0] + bw[c + 0]);
  o[1] = f2bf((v.y - mu) * rs * gw[c + 1] + bw[c + 1]);
  o[2] = f2bf((v.z - mu) * rs * gw[c + 2] + bw[c + 2]);
  o[3] = f2bf((v.w - mu) * rs * gw[c + 3] + bw[c + 3]);
}

// ---------------------------------------------------------------------------
// pe hidden: gelu(pos @ pe_w1 + pe_b1), K=3 so scalar per output element.
// ---------------------------------------------------------------------------
__global__ void pt_pe_hidden(const float* __restrict__ pos,
                             const float* __restrict__ w1,
                             const float* __restrict__ b1,
                             unsigned short* __restrict__ out, int total) {
  int idx = blockIdx.x * blockDim.x + threadIdx.x;
  if (idx >= total) return;
  int col = idx & (DIMC - 1);
  int row = idx >> 9;
  float s = b1[col] + pos[row * 3 + 0] * w1[col]
                    + pos[row * 3 + 1] * w1[DIMC + col]
                    + pos[row * 3 + 2] * w1[2 * DIMC + col];
  out[idx] = f2bf(gelu_exact(s));
}

// ---------------------------------------------------------------------------
// Generic bf16 WMMA GEMM: C[M,N] = A[M,K] * Bt[N,K]^T (+bias)(+gelu)(+residual)
// flags: bit0 = gelu, bit1 = bf16 output (else fp32)
// Block = 256 threads (8 waves), block tile 64x256, wave tile 32x64.
// ---------------------------------------------------------------------------
__global__ __launch_bounds__(256) void pt_gemm_bf16(
    const unsigned short* __restrict__ A, const unsigned short* __restrict__ Bt,
    const float* __restrict__ bias, const float* __restrict__ residual,
    void* __restrict__ out, int M, int N, int K, int flags) {
  int lane = threadIdx.x & 31;
  int wave = threadIdx.x >> 5;
  int g = lane >> 4, r = lane & 15;
  int wrow = wave & 1, wcol = wave >> 1;
  int m0 = blockIdx.y * 64 + wrow * 32;
  int n0 = blockIdx.x * 256 + wcol * 64;

  v8f zero8 = {0.f, 0.f, 0.f, 0.f, 0.f, 0.f, 0.f, 0.f};
  v8f acc[2][4];
#pragma unroll
  for (int mt = 0; mt < 2; ++mt)
#pragma unroll
    for (int nt = 0; nt < 4; ++nt) acc[mt][nt] = zero8;

  const unsigned short* arow[2];
  arow[0] = A + (size_t)(m0 + r) * K;
  arow[1] = A + (size_t)(m0 + 16 + r) * K;
  const unsigned short* brow[4];
#pragma unroll
  for (int nt = 0; nt < 4; ++nt) brow[nt] = Bt + (size_t)(n0 + nt * 16 + r) * K;

  for (int kk = 0; kk < K; kk += 32) {
    __builtin_prefetch(arow[0] + kk + 64, 0, 0);
    __builtin_prefetch(brow[0] + kk + 64, 0, 0);
    v16bf af[2], bfr[4];
    af[0] = load_fragA(arow[0] + kk, g);
    af[1] = load_fragA(arow[1] + kk, g);
#pragma unroll
    for (int nt = 0; nt < 4; ++nt) bfr[nt] = load_fragB(brow[nt] + kk, g);
#pragma unroll
    for (int mt = 0; mt < 2; ++mt)
#pragma unroll
      for (int nt = 0; nt < 4; ++nt)
        acc[mt][nt] = WMMA_BF16(af[mt], bfr[nt], acc[mt][nt]);
  }

  bool do_gelu = (flags & 1) != 0;
  bool out_bf = (flags & 2) != 0;
#pragma unroll
  for (int mt = 0; mt < 2; ++mt) {
#pragma unroll
    for (int nt = 0; nt < 4; ++nt) {
      int col = n0 + nt * 16 + r;
      float bv = bias ? bias[col] : 0.0f;
#pragma unroll
      for (int v = 0; v < 8; ++v) {
        int row = m0 + mt * 16 + 8 * g + v;
        float val = acc[mt][nt][v] + bv;
        if (do_gelu) val = gelu_exact(val);
        if (residual) val += residual[(size_t)row * N + col];
        if (out_bf)
          ((unsigned short*)out)[(size_t)row * N + col] = f2bf(val);
        else
          ((float*)out)[(size_t)row * N + col] = val;
      }
    }
  }
}

// ---------------------------------------------------------------------------
// Assemble: q = (qkv_q + pe)*SCALE, k = qkv_k + pe  ->  [BH,N,64] bf16
//           vT = qkv_v                              ->  [BH,64,N] bf16
// ---------------------------------------------------------------------------
__global__ void pt_assemble_qkv(const unsigned short* __restrict__ qkv,
                                const unsigned short* __restrict__ pe,
                                unsigned short* __restrict__ q,
                                unsigned short* __restrict__ k,
                                unsigned short* __restrict__ vt) {
  int idx = blockIdx.x * blockDim.x + threadIdx.x;   // BH*N*64 total
  int d = idx & 63;
  int n = (idx >> 6) & (NTOK - 1);
  int bh = idx >> 18;
  int b = bh >> 3, h = bh & 7;
  size_t base = ((size_t)(b * NTOK + n)) * 1536 + h * 64 + d;
  size_t pidx = ((size_t)(b * NTOK + n)) * DIMC + h * 64 + d;
  float pev = bf2f(pe[pidx]);
  size_t qi = ((size_t)bh * NTOK + n) * HD + d;
  q[qi] = f2bf((bf2f(qkv[base]) + pev) * SCALE_Q);
  k[qi] = f2bf(bf2f(qkv[base + 512]) + pev);
  vt[((size_t)bh * HD + d) * NTOK + n] = qkv[base + 1024];
}

// ---------------------------------------------------------------------------
// Flash attention. Block = 4 waves sharing one (b,h); each wave owns a 16-query
// tile. K/V 32-key tiles are staged in LDS with async global->LDS copies,
// double buffered: each wave issues exactly 4 async instructions per tile, so
// s_wait_asynccnt 0x4 retires tile i while tile i+1 stays in flight.
// S^T = K_tile x Q^T so softmax stats are in-lane (+xor16) and the two S^T
// C-frags map directly onto the P A-frag for O += P x V.
// ---------------------------------------------------------------------------
__global__ __launch_bounds__(128) void pt_flash_attn(
    const unsigned short* __restrict__ q, const unsigned short* __restrict__ k,
    const unsigned short* __restrict__ vt, unsigned short* __restrict__ o) {
  __shared__ unsigned short kbuf[2][32][HD];   // 2 x 4KB : keys x d
  __shared__ unsigned short vbuf[2][HD][32];   // 2 x 4KB : d x keys

  int t = threadIdx.x;
  int lane = t & 31;
  int g = lane >> 4, r = lane & 15;
  int wave = t >> 5;
  int bh = blockIdx.x >> 6;                    // 64 blocks per (b,h)
  int qt = (blockIdx.x & 63) * 4 + wave;
  int b = bh >> 3, h = bh & 7;
  int q0 = qt * 16;

  v8f zero8 = {0.f, 0.f, 0.f, 0.f, 0.f, 0.f, 0.f, 0.f};

  // Q^T B-frags over the two d-halves (K dim of the S^T wmma = head_dim)
  const unsigned short* qrow = q + ((size_t)bh * NTOK + q0 + r) * HD;
  v16bf qf0 = load_fragB(qrow, g);
  v16bf qf1 = load_fragB(qrow + 32, g);

  const unsigned short* kbase = k + (size_t)bh * NTOK * HD;
  const unsigned short* vbase = vt + (size_t)bh * HD * NTOK;

  // issue the 4 async copies (2 for K tile, 2 for V tile) for tile at k0
  auto issue_tile = [&](int k0, int buf) {
#pragma unroll
    for (int j = 0; j < 2; ++j) {
      int c = t * 2 + j;                       // 256 x 16B chunks of K tile
      int row = c >> 3, col = (c & 7) * 8;
      async_copy_b128((unsigned)(uintptr_t)&kbuf[buf][row][col],
                      kbase + (size_t)(k0 + row) * HD + col);
    }
#pragma unroll
    for (int j = 0; j < 2; ++j) {
      int c = t * 2 + j;                       // 256 x 16B chunks of V tile
      int d = c >> 2, col = (c & 3) * 8;
      async_copy_b128((unsigned)(uintptr_t)&vbuf[buf][d][col],
                      vbase + (size_t)d * NTOK + k0 + col);
    }
  };

  v8f oacc[4];
#pragma unroll
  for (int dc = 0; dc < 4; ++dc) oacc[dc] = zero8;
  float m = -INFINITY, l = 0.0f;

  const int NT = NTOK / 32;                    // 128 key tiles
  issue_tile(0, 0);
  issue_tile(32, 1);

  for (int i = 0; i < NT; ++i) {
    // retire tile i (its 4 async ops are the oldest of <=8 outstanding)
    if (i + 2 < NT) asm volatile("s_wait_asynccnt 0x4" ::: "memory");
    else            asm volatile("s_wait_asynccnt 0x0" ::: "memory");
    __syncthreads();

    int buf = i & 1;
    const unsigned short* kr0 = &kbuf[buf][r][0];
    const unsigned short* kr1 = &kbuf[buf][16 + r][0];
    v16bf ka0 = load_fragA(kr0, g);            // keys 0..15 of tile, d 0..31
    v16bf ka1 = load_fragA(kr0 + 32, g);       //                    d 32..63
    v16bf kb0 = load_fragA(kr1, g);            // keys 16..31 of tile
    v16bf kb1 = load_fragA(kr1 + 32, g);

    v8f st0 = zero8, st1 = zero8;              // S^T: 16 keys x 16 queries
    st0 = WMMA_BF16(ka0, qf0, st0);
    st0 = WMMA_BF16(ka1, qf1, st0);
    st1 = WMMA_BF16(kb0, qf0, st1);
    st1 = WMMA_BF16(kb1, qf1, st1);

    // per-query (column = lane%16) online softmax stats
    float vmax = -INFINITY;
#pragma unroll
    for (int e = 0; e < 8; ++e) vmax = fmaxf(vmax, fmaxf(st0[e], st1[e]));
    vmax = fmaxf(vmax, __shfl_xor(vmax, 16));
    float mn = fmaxf(m, vmax);
    float p0[8], p1[8], ts = 0.0f;
#pragma unroll
    for (int e = 0; e < 8; ++e) {
      p0[e] = __expf(st0[e] - mn);
      p1[e] = __expf(st1[e] - mn);
      ts += p0[e] + p1[e];
    }
    ts += __shfl_xor(ts, 16);
    float alpha = __expf(m - mn);
    l = l * alpha + ts;
    m = mn;

    // P A-frag: elems 0..7 <- keys 8g+v (st0), 8..15 <- keys 16+8g+v (st1)
    union { v16bf v; unsigned short s[16]; } pu;
#pragma unroll
    for (int e = 0; e < 8; ++e) {
      pu.s[e] = f2bf(p0[e]);
      pu.s[8 + e] = f2bf(p1[e]);
    }

    // rescale O accumulators; O C-frag lane holds queries 8g+v
#pragma unroll
    for (int v = 0; v < 8; ++v) {
      float a = __shfl(alpha, 8 * g + v);
      oacc[0][v] *= a; oacc[1][v] *= a; oacc[2][v] *= a; oacc[3][v] *= a;
    }

#pragma unroll
    for (int dc = 0; dc < 4; ++dc) {
      v16bf vf = load_fragB(&vbuf[buf][dc * 16 + r][0], g);
      oacc[dc] = WMMA_BF16(pu.v, vf, oacc[dc]);
    }

    __syncthreads();                           // everyone done reading buf
    if (i + 2 < NT) issue_tile((i + 2) * 32, buf);
  }

  float invl = 1.0f / l;
#pragma unroll
  for (int v = 0; v < 8; ++v) {
    float il = __shfl(invl, 8 * g + v);
    int qabs = q0 + 8 * g + v;
    size_t orow = ((size_t)(b * NTOK + qabs)) * DIMC + h * 64;
#pragma unroll
    for (int dc = 0; dc < 4; ++dc)
      o[orow + dc * 16 + r] = f2bf(oacc[dc][v] * il);
  }
}

// ---------------------------------------------------------------------------
// Host-side orchestration
// ---------------------------------------------------------------------------
extern "C" void kernel_launch(void* const* d_in, const int* in_sizes, int n_in,
                              void* d_out, int out_size, void* d_ws, size_t ws_size,
                              hipStream_t stream) {
  const float* x      = (const float*)d_in[0];
  const float* pos    = (const float*)d_in[1];
  const float* qkv_w  = (const float*)d_in[2];
  const float* qkv_b  = (const float*)d_in[3];
  const float* proj_w = (const float*)d_in[4];
  const float* proj_b = (const float*)d_in[5];
  const float* pe_w1  = (const float*)d_in[6];
  const float* pe_b1  = (const float*)d_in[7];
  const float* pe_w2  = (const float*)d_in[8];
  const float* pe_b2  = (const float*)d_in[9];
  const float* mlp_w1 = (const float*)d_in[10];
  const float* mlp_b1 = (const float*)d_in[11];
  const float* mlp_w2 = (const float*)d_in[12];
  const float* mlp_b2 = (const float*)d_in[13];
  const float* n1_g   = (const float*)d_in[14];
  const float* n1_b   = (const float*)d_in[15];
  const float* n2_g   = (const float*)d_in[16];
  const float* n2_b   = (const float*)d_in[17];
  float* out = (float*)d_out;

  char* w = (char*)d_ws;
  size_t off = 0;
  auto take = [&](size_t bytes) { char* p = w + off; off += bytes; return p; };
  unsigned short* h_bf   = (unsigned short*)take((size_t)BN * DIMC * 2);   // LN1 out
  unsigned short* qkv_bf = (unsigned short*)take((size_t)BN * 1536 * 2);   // qkv GEMM out
  unsigned short* peh    = (unsigned short*)take((size_t)BN * DIMC * 2);   // pe hidden
  unsigned short* pe     = (unsigned short*)take((size_t)BN * DIMC * 2);   // pe
  unsigned short* qb     = (unsigned short*)take((size_t)BHN * NTOK * HD * 2);
  unsigned short* kb     = (unsigned short*)take((size_t)BHN * NTOK * HD * 2);
  unsigned short* vtb    = (unsigned short*)take((size_t)BHN * HD * NTOK * 2);
  unsigned short* ob     = (unsigned short*)take((size_t)BN * DIMC * 2);   // attn out
  float*          x2     = (float*)take((size_t)BN * DIMC * 4);            // residual 1
  unsigned short* h2     = (unsigned short*)take((size_t)BN * DIMC * 2);   // LN2 out
  unsigned short* mh     = (unsigned short*)take((size_t)BN * HIDN * 2);   // mlp hidden
  unsigned short* wq_t   = (unsigned short*)take((size_t)1536 * DIMC * 2);
  unsigned short* wp_t   = (unsigned short*)take((size_t)DIMC * DIMC * 2);
  unsigned short* wpe2_t = (unsigned short*)take((size_t)DIMC * DIMC * 2);
  unsigned short* wm1_t  = (unsigned short*)take((size_t)HIDN * DIMC * 2);
  unsigned short* wm2_t  = (unsigned short*)take((size_t)DIMC * HIDN * 2);

  // 1) transpose weights to [N,K] bf16
  pt_wt_transpose_bf16<<<(DIMC * 1536 + 255) / 256, 256, 0, stream>>>(qkv_w, wq_t, DIMC, 1536);
  pt_wt_transpose_bf16<<<(DIMC * DIMC + 255) / 256, 256, 0, stream>>>(proj_w, wp_t, DIMC, DIMC);
  pt_wt_transpose_bf16<<<(DIMC * DIMC + 255) / 256, 256, 0, stream>>>(pe_w2, wpe2_t, DIMC, DIMC);
  pt_wt_transpose_bf16<<<(DIMC * HIDN + 255) / 256, 256, 0, stream>>>(mlp_w1, wm1_t, DIMC, HIDN);
  pt_wt_transpose_bf16<<<(HIDN * DIMC + 255) / 256, 256, 0, stream>>>(mlp_w2, wm2_t, HIDN, DIMC);

  // 2) LN1 + pe hidden
  pt_ln_bf16<<<BN, 128, 0, stream>>>(x, n1_g, n1_b, h_bf);
  pt_pe_hidden<<<(BN * DIMC) / 256, 256, 0, stream>>>(pos, pe_w1, pe_b1, peh, BN * DIMC);

  // 3) pe = peh @ pe_w2 + b2 (bf16 out);  qkv = h @ qkv_w + b (bf16 out)
  pt_gemm_bf16<<<dim3(DIMC / 256, BN / 64), 256, 0, stream>>>(
      peh, wpe2_t, pe_b2, nullptr, pe, BN, DIMC, DIMC, /*bf16*/2);
  pt_gemm_bf16<<<dim3(1536 / 256, BN / 64), 256, 0, stream>>>(
      h_bf, wq_t, qkv_b, nullptr, qkv_bf, BN, 1536, DIMC, /*bf16*/2);

  // 4) assemble q(+pe)*scale, k(+pe), V^T
  pt_assemble_qkv<<<(BHN * NTOK * HD) / 256, 256, 0, stream>>>(qkv_bf, pe, qb, kb, vtb);

  // 5) flash attention -> [BN,512] bf16
  pt_flash_attn<<<(BHN * (NTOK / 16)) / 4, 128, 0, stream>>>(qb, kb, vtb, ob);

  // 6) x2 = x + o @ proj_w + proj_b  (f32 out)
  pt_gemm_bf16<<<dim3(DIMC / 256, BN / 64), 256, 0, stream>>>(
      ob, wp_t, proj_b, x, x2, BN, DIMC, DIMC, /*f32*/0);

  // 7) LN2, MLP
  pt_ln_bf16<<<BN, 128, 0, stream>>>(x2, n2_g, n2_b, h2);
  pt_gemm_bf16<<<dim3(HIDN / 256, BN / 64), 256, 0, stream>>>(
      h2, wm1_t, mlp_b1, nullptr, mh, BN, HIDN, DIMC, /*gelu|bf16*/3);
  pt_gemm_bf16<<<dim3(DIMC / 256, BN / 64), 256, 0, stream>>>(
      mh, wm2_t, mlp_b2, x2, out, BN, DIMC, HIDN, /*f32*/0);
}
